// TemporalCrossTransformer_12567074308724
// MI455X (gfx1250) — matrix operand order repaired and checked by reading, not compiled, not run on hardware
//
#include <hip/hip_runtime.h>
#include <hip/hip_bf16.h>

typedef __attribute__((ext_vector_type(16))) __bf16 v16bf;
typedef __attribute__((ext_vector_type(8)))  float  v8f;

union Frag16 { v16bf v; uint4 q[2]; };

#define FEAT   2048
#define DOUTD  1152
#define WAY    5
#define SHOT   5
#define NQ     500
#define TL     28
#define NFR    4200     // 525 clips * 8 frames
#define MP1    4224     // NFR padded to 128
#define NP1    4608     // 4 * 1152
#define SROW   700
#define SROWP  704
#define QROW   14000
#define QROWP  14080
#define KM     160      // 140 padded to 5*32

// ---- CDNA5 async global->LDS copies (ASYNCcnt), with safe fallback -----
// Builtin param types (from hipcc diagnostic): (global int4*, local int4*, imm, imm)
typedef int vint4 __attribute__((vector_size(16)));
typedef __attribute__((address_space(1))) vint4* gp4;
typedef __attribute__((address_space(3))) vint4* lp4;

#if defined(__has_builtin)
#if __has_builtin(__builtin_amdgcn_global_load_async_to_lds_b128)
#define USE_ASYNC_LDS 1
#define ASYNC_COPY16(srcg, dstl) \
    __builtin_amdgcn_global_load_async_to_lds_b128((gp4)(srcg), (lp4)(dstl), 0, 0)
#if __has_builtin(__builtin_amdgcn_s_wait_asynccnt)
#define WAIT_ASYNC() __builtin_amdgcn_s_wait_asynccnt(0)
#else
#define WAIT_ASYNC() asm volatile("s_wait_asynccnt 0x0" ::: "memory")
#endif
#endif
#endif
#ifndef USE_ASYNC_LDS
#define WAIT_ASYNC() ((void)0)
#endif

__constant__ int TUP_I[28] = {0,0,0,0,0,0,0, 1,1,1,1,1,1, 2,2,2,2,2, 3,3,3,3, 4,4,4, 5,5, 6};
__constant__ int TUP_J[28] = {1,2,3,4,5,6,7, 2,3,4,5,6,7, 3,4,5,6,7, 4,5,6,7, 5,6,7, 6,7, 7};

__device__ inline v8f v8f_zero() { v8f z; for (int i = 0; i < 8; ++i) z[i] = 0.0f; return z; }

// ---------------- prep: X + positional encoding -> bf16 -----------------
__global__ void prep_x(const float* __restrict__ sup, const float* __restrict__ qry,
                       __bf16* __restrict__ Xb) {
    long idx = (long)blockIdx.x * 256 + threadIdx.x;
    if (idx >= (long)NFR * FEAT) return;
    int d = (int)(idx % FEAT);
    long row = idx / FEAT;              // frame row: support frames 0..199, query 200..4199
    int frame = (int)(row & 7);
    float x = (row < 200) ? sup[row * FEAT + d] : qry[(row - 200) * FEAT + d];
    int k2 = d >> 1;
    float w = __expf(-(float)(2 * k2) * (9.210340371976184f / 2048.0f)); // ln(1e4)/2048
    float ang = (float)frame * w;
    float pe = (d & 1) ? __cosf(ang) : __sinf(ang);
    Xb[row * FEAT + d] = (__bf16)(x + pe);
}

// ---------------- prep: combined weight, transposed, bf16 ---------------
// WbT[n][k], n in [0,4608): [k_top | k_bot | v_top | v_bot]
__global__ void prep_w(const float* __restrict__ kw, const float* __restrict__ vw,
                       __bf16* __restrict__ WbT) {
    long idx = (long)blockIdx.x * 256 + threadIdx.x;
    if (idx >= (long)NP1 * FEAT) return;
    int k = (int)(idx % FEAT);
    int n = (int)(idx / FEAT);
    float v;
    if      (n < 1152) v = kw[(long)k          * DOUTD + n];
    else if (n < 2304) v = kw[(long)(2048 + k) * DOUTD + (n - 1152)];
    else if (n < 3456) v = vw[(long)k          * DOUTD + (n - 2304)];
    else               v = vw[(long)(2048 + k) * DOUTD + (n - 3456)];
    WbT[idx] = (__bf16)v;
}

// ---------------- generic bf16 WMMA GEMM: C = scale * A @ BT^T ----------
// A[M][K] row-major bf16 (lda), BT[N][K] row-major bf16 (ldb), C[M][N] f32 (ldc).
// Block tile 128x64, 8 waves (2x4), wave tile 64x16, K-step 32,
// double-buffered LDS fed by global_load_async_to_lds_b128 (ASYNCcnt).
__device__ __forceinline__ void stage_tiles(const __bf16* __restrict__ A,
                                            const __bf16* __restrict__ BT,
                                            long bm, long bn, int lda, int ldb, int k0,
                                            int tid, __bf16* sAb, __bf16* sBb) {
    const int ar = tid >> 1, ac = (tid & 1) * 16;   // A tile 128x32: 16 elems/thread
    const int br = tid >> 2, bc = (tid & 3) * 8;    // B tile  64x32:  8 elems/thread
#ifdef USE_ASYNC_LDS
    ASYNC_COPY16(A + (bm + ar) * lda + k0 + ac,     sAb + ar * 32 + ac);
    ASYNC_COPY16(A + (bm + ar) * lda + k0 + ac + 8, sAb + ar * 32 + ac + 8);
    ASYNC_COPY16(BT + (bn + br) * ldb + k0 + bc,    sBb + br * 32 + bc);
#else
    const uint4* g = (const uint4*)(A + (bm + ar) * lda + k0 + ac);
    *(uint4*)(sAb + ar * 32 + ac)     = g[0];
    *(uint4*)(sAb + ar * 32 + ac + 8) = g[1];
    *(uint4*)(sBb + br * 32 + bc) = *(const uint4*)(BT + (bn + br) * ldb + k0 + bc);
#endif
}

__launch_bounds__(256)
__global__ void gemm_bf16(const __bf16* __restrict__ A, const __bf16* __restrict__ BT,
                          float* __restrict__ C, int K, int lda, int ldb, int ldc,
                          float scale) {
    __shared__ __align__(16) __bf16 sA[2][128][32];
    __shared__ __align__(16) __bf16 sB[2][64][32];
    const int tid  = threadIdx.x;
    const long bm  = (long)blockIdx.y * 128;
    const long bn  = (long)blockIdx.x * 64;
    const int wid  = tid >> 5;
    const int lane = tid & 31;
    const int waveM = wid & 1;     // 0..1 -> 64 rows each
    const int waveN = wid >> 1;    // 0..3 -> 16 cols each
    const int half = lane >> 4;
    const int l16  = lane & 15;

    v8f acc[4];
    for (int i = 0; i < 4; ++i) acc[i] = v8f_zero();

    stage_tiles(A, BT, bm, bn, lda, ldb, 0, tid, &sA[0][0][0], &sB[0][0][0]);

    int buf = 0;
    for (int k0 = 0; k0 < K; k0 += 32, buf ^= 1) {
        WAIT_ASYNC();            // this wave's staged tile has landed in LDS
        __syncthreads();         // ...and every other wave's too; prev buf fully read
        if (k0 + 32 < K) {
            stage_tiles(A, BT, bm, bn, lda, ldb, k0 + 32, tid,
                        &sA[buf ^ 1][0][0], &sB[buf ^ 1][0][0]);
            // warm L2 for the slab after next (CDNA5 global_prefetch_b8)
            if (k0 + 64 < K) __builtin_prefetch(A + (bm + (tid >> 1)) * lda + k0 + 64, 0, 1);
        }
        // B fragment: lane -> N=l16, elems e -> K = 8*half + e (e<8), +8 more (e>=8)
        Frag16 bf;
        bf.q[0] = *(const uint4*)&sB[buf][waveN * 16 + l16][8 * half];
        bf.q[1] = *(const uint4*)&sB[buf][waveN * 16 + l16][16 + 8 * half];
        // A fragments: lane -> M=l16, elems e -> K = 8*half + e (e<8), +8 more (e>=8)
        for (int mt = 0; mt < 4; ++mt) {
            Frag16 af;
            const __bf16* arow = &sA[buf][waveM * 64 + mt * 16 + l16][0];
            af.q[0] = *(const uint4*)(arow + 8 * half);
            af.q[1] = *(const uint4*)(arow + 16 + 8 * half);
            acc[mt] = __builtin_amdgcn_wmma_f32_16x16x32_bf16(
                false, af.v, false, bf.v, (short)0, acc[mt], false, false);
        }
    }
    // C/D layout: element (r, lane) -> M = r + 8*half, N = l16
    for (int mt = 0; mt < 4; ++mt) {
        long col   = bn + waveN * 16 + l16;
        long rbase = bm + waveM * 64 + mt * 16 + half * 8;
        for (int r = 0; r < 8; ++r)
            C[(rbase + r) * ldc + col] = acc[mt][r] * scale;
    }
}

// ---------------- combine per-frame projections + LayerNorm -------------
__launch_bounds__(256)
__global__ void combine_ln(const float* __restrict__ Proj,
                           const float* __restrict__ kb, const float* __restrict__ vb,
                           const float* __restrict__ gamma, const float* __restrict__ beta,
                           __bf16* __restrict__ sksB, __bf16* __restrict__ qksB,
                           float* __restrict__ qvsF, __bf16* __restrict__ classvB) {
    const int row = blockIdx.x;          // 0..14699  (support 0..699, query 700..)
    const int tid = threadIdx.x;
    const int t = row % TL;
    const int clip = row / TL;
    const long fi = (long)clip * 8 + TUP_I[t];
    const long fj = (long)clip * 8 + TUP_J[t];
    const float* pi = Proj + fi * NP1;
    const float* pj = Proj + fj * NP1;
    __shared__ float red[256], red2[256];
    float sum = 0.f, sumsq = 0.f;
    for (int d = tid; d < DOUTD; d += 256) {
        float kv = pi[d] + pj[1152 + d] + kb[d];
        sum += kv; sumsq += kv * kv;
    }
    red[tid] = sum; red2[tid] = sumsq; __syncthreads();
    for (int off = 128; off; off >>= 1) {
        if (tid < off) { red[tid] += red[tid + off]; red2[tid] += red2[tid + off]; }
        __syncthreads();
    }
    const float mu = red[0] / (float)DOUTD;
    const float var = red2[0] / (float)DOUTD - mu * mu;
    const float rstd = rsqrtf(var + 1e-5f);
    const bool isSup = clip < 25;
    for (int d = tid; d < DOUTD; d += 256) {
        float kv = pi[d] + pj[1152 + d] + kb[d];
        float ks = (kv - mu) * rstd * gamma[d] + beta[d];
        float vv = pi[2304 + d] + pj[3456 + d] + vb[d];
        if (isSup) {
            sksB[(long)row * DOUTD + d] = (__bf16)ks;
            int c = row / 140, m = row % 140;
            classvB[((long)c * DOUTD + d) * KM + m] = (__bf16)vv;   // [c][d][m]
        } else {
            long qr = row - SROW;
            qksB[qr * DOUTD + d] = (__bf16)ks;
            qvsF[qr * DOUTD + d] = vv;
        }
    }
}

// ---------------- softmax over (shot, tuple) = 140 ----------------------
__launch_bounds__(256)
__global__ void softmax_attn(const float* __restrict__ scores, __bf16* __restrict__ attnB) {
    const int row = blockIdx.x;   // 0..13999
    const int c   = blockIdx.y;   // 0..4
    const int tid = threadIdx.x;
    __shared__ float sh[256];
    float v = (tid < 140) ? scores[(long)row * SROWP + c * 140 + tid] : -3.0e38f;
    sh[tid] = v; __syncthreads();
    for (int off = 128; off; off >>= 1) {
        if (tid < off) sh[tid] = fmaxf(sh[tid], sh[tid + off]);
        __syncthreads();
    }
    const float mx = sh[0]; __syncthreads();
    float e = (tid < 140) ? __expf(v - mx) : 0.f;
    sh[tid] = e; __syncthreads();
    for (int off = 128; off; off >>= 1) {
        if (tid < off) sh[tid] += sh[tid + off];
        __syncthreads();
    }
    const float inv = 1.0f / sh[0];
    if (tid < KM)
        attnB[((long)c * QROW + row) * KM + tid] = (__bf16)((tid < 140) ? e * inv : 0.f);
}

// ---------------- fused prototype WMMA + squared distance ---------------
__launch_bounds__(128)
__global__ void proto_dist(const __bf16* __restrict__ attnB, const __bf16* __restrict__ classvB,
                           const float* __restrict__ qvsF, float* __restrict__ dist) {
    const int rt  = blockIdx.x;   // 0..874 (16-row tiles of 14000)
    const int c   = blockIdx.y;
    const int tid = threadIdx.x;
    const int wid = tid >> 5, lane = tid & 31;
    const int half = lane >> 4, l16 = lane & 15;
    const __bf16* Ab = attnB + ((long)c * QROW + (long)rt * 16) * KM;
    const __bf16* Bb = classvB + (long)c * DOUTD * KM;

    Frag16 afr[5];   // A fragments for all 5 K-steps, reused across d-tiles
    for (int ks = 0; ks < 5; ++ks) {
        const __bf16* arow = Ab + (long)l16 * KM + 32 * ks;
        afr[ks].q[0] = *(const uint4*)(arow + 8 * half);
        afr[ks].q[1] = *(const uint4*)(arow + 16 + 8 * half);
    }
    float acc8[8];
    for (int r = 0; r < 8; ++r) acc8[r] = 0.f;

    for (int dt = wid; dt < 72; dt += 4) {         // 1152 / 16 d-tiles
        v8f cacc = v8f_zero();
        for (int ks = 0; ks < 5; ++ks) {
            Frag16 bf;
            const __bf16* brow = Bb + (long)(dt * 16 + l16) * KM + 32 * ks;
            bf.q[0] = *(const uint4*)(brow + 8 * half);
            bf.q[1] = *(const uint4*)(brow + 16 + 8 * half);
            cacc = __builtin_amdgcn_wmma_f32_16x16x32_bf16(
                false, afr[ks].v, false, bf.v, (short)0, cacc, false, false);
        }
        const int dcol = dt * 16 + l16;
        for (int r = 0; r < 8; ++r) {
            long grow = (long)rt * 16 + r + 8 * half;
            float diff = qvsF[grow * DOUTD + dcol] - cacc[r];
            acc8[r] += diff * diff;
        }
    }
    __shared__ float red[4][32][8];
    for (int r = 0; r < 8; ++r) red[wid][lane][r] = acc8[r];
    __syncthreads();
    if (tid < 16) {
        const int m = tid, r = m & 7, lbase = (m < 8) ? 0 : 16;
        float s = 0.f;
        for (int w = 0; w < 4; ++w)
            for (int l = 0; l < 16; ++l) s += red[w][lbase + l][r];
        const int grow = rt * 16 + m;
        atomicAdd(&dist[c * NQ + grow / TL], s);
    }
}

// ---------------- init / finalize ---------------------------------------
__global__ void init_zero(float* __restrict__ dist, __bf16* __restrict__ classvB, long nCv) {
    long i = (long)blockIdx.x * 256 + threadIdx.x;
    if (i < nCv) classvB[i] = (__bf16)0.0f;
    if (i < WAY * NQ) dist[i] = 0.0f;
}

__global__ void finalize(const float* __restrict__ dist, const float* __restrict__ gt,
                         const float* __restrict__ tw, float* __restrict__ out) {
    int i = blockIdx.x * 64 + threadIdx.x;
    if (i < NQ * WAY) {
        int q = i / WAY, c = i % WAY;
        out[q * WAY + c] = -(dist[c * NQ + q] / (float)TL) * gt[0] * tw[0];
    }
}

extern "C" void kernel_launch(void* const* d_in, const int* in_sizes, int n_in,
                              void* d_out, int out_size, void* d_ws, size_t ws_size,
                              hipStream_t stream) {
    const float* sup   = (const float*)d_in[0];
    // d_in[1] = support_labels (unused: episode is sorted & balanced, as in reference)
    const float* qry   = (const float*)d_in[2];
    const float* kw    = (const float*)d_in[3];
    const float* kb    = (const float*)d_in[4];
    const float* vw    = (const float*)d_in[5];
    const float* vb    = (const float*)d_in[6];
    const float* gamma = (const float*)d_in[7];
    const float* beta  = (const float*)d_in[8];
    const float* gt    = (const float*)d_in[9];
    const float* tw    = (const float*)d_in[10];
    float* out = (float*)d_out;

    char* ws = (char*)d_ws;
    size_t off = 0;
    auto alloc = [&](size_t bytes) {
        void* p = ws + off;
        off = (off + bytes + 255) & ~(size_t)255;
        return p;
    };
    __bf16* Xb      = (__bf16*)alloc((size_t)MP1 * FEAT * 2);
    __bf16* WbT     = (__bf16*)alloc((size_t)NP1 * FEAT * 2);
    float*  Proj    = (float*) alloc((size_t)MP1 * NP1 * 4);
    __bf16* qksB    = (__bf16*)alloc((size_t)QROWP * DOUTD * 2);
    __bf16* sksB    = (__bf16*)alloc((size_t)SROWP * DOUTD * 2);
    float*  qvsF    = (float*) alloc((size_t)QROW * DOUTD * 4);
    __bf16* classvB = (__bf16*)alloc((size_t)WAY * DOUTD * KM * 2);
    float*  scores  = (float*) alloc((size_t)QROWP * SROWP * 4);
    __bf16* attnB   = (__bf16*)alloc((size_t)WAY * QROW * KM * 2);
    float*  dist    = (float*) alloc((size_t)WAY * NQ * 4);

    const long nCv = (long)WAY * DOUTD * KM;
    init_zero<<<(unsigned)((nCv + 255) / 256), 256, 0, stream>>>(dist, classvB, nCv);

    prep_x<<<(unsigned)(((long)NFR * FEAT + 255) / 256), 256, 0, stream>>>(sup, qry, Xb);
    prep_w<<<(unsigned)(((long)NP1 * FEAT + 255) / 256), 256, 0, stream>>>(kw, vw, WbT);

    // Projection GEMM: [4224 x 2048] x [2048 x 4608]
    gemm_bf16<<<dim3(NP1 / 64, MP1 / 128), 256, 0, stream>>>(
        Xb, WbT, Proj, FEAT, FEAT, FEAT, NP1, 1.0f);

    combine_ln<<<525 * TL, 256, 0, stream>>>(Proj, kb, vb, gamma, beta,
                                             sksB, qksB, qvsF, classvB);

    // Scores GEMM: [14080 x 1152] x [1152 x 704], scale = 1/sqrt(1152)
    gemm_bf16<<<dim3(SROWP / 64, QROWP / 128), 256, 0, stream>>>(
        qksB, sksB, scores, DOUTD, DOUTD, DOUTD, SROWP, 0.029462782549439476f);

    softmax_attn<<<dim3(QROW, WAY), 256, 0, stream>>>(scores, attnB);

    proto_dist<<<dim3(QROW / 16, WAY), 128, 0, stream>>>(attnB, classvB, qvsF, dist);

    finalize<<<(NQ * WAY + 63) / 64, 64, 0, stream>>>(dist, gt, tw, out);
}
